// L2MLoRAqkv_3805341024603
// MI455X (gfx1250) — compile-verified, matrix operand order checked by reading, not run on hardware
//
#include <hip/hip_runtime.h>

typedef __bf16 bf16_t;
typedef __bf16 v16bf __attribute__((ext_vector_type(16)));
typedef __bf16 v8bf  __attribute__((ext_vector_type(8)));
typedef float  v8f   __attribute__((ext_vector_type(8)));

#define DIM   1024
#define ODIM  3072
#define NROW  4096
#define BATCH 8
#define RANK  8
#define TM    128
#define TO    128
#define TK    32
#define NIT   (DIM / TK)
#define LDST  40   // LDS row stride in bf16 elems (32 + 8 pad -> 80B rows, bank spread)

// ---------------------------------------------------------------------------
// CDNA5 helpers: async global->LDS DMA (ASYNCcnt) + wait.
// LDS dest address = low 32 bits of the generic pointer (ISA: LDS aperture
// truncation), global address = SGPR64 base + per-lane 32-bit byte offset.
// ---------------------------------------------------------------------------
__device__ __forceinline__ uint32_t lds_off(const void* p) {
    return (uint32_t)(uintptr_t)p;
}
__device__ __forceinline__ void async_ld_b128(uint32_t ldsaddr, uint64_t sbase, uint32_t voff) {
    asm volatile("global_load_async_to_lds_b128 %0, %1, %2"
                 :
                 : "v"(ldsaddr), "v"(voff), "s"(sbase)
                 : "memory");
}
__device__ __forceinline__ void wait_async0() {
    asm volatile("s_wait_asynccnt 0x0" ::: "memory");
}

// ---------------------------------------------------------------------------
// Kernel 1: fold LoRA (A@B) into the base weight per batch, emit bf16 hi/lo.
// W_eff[b][o][i] = W[o][i] + sum_r A[i][r]*B[r][o]   (q rows o<1024, v rows o>=2048)
// ---------------------------------------------------------------------------
__global__ void __launch_bounds__(256) merge_w_kernel(
    const float* __restrict__ weight,
    const float* __restrict__ Aq, const float* __restrict__ Bq,
    const float* __restrict__ Av, const float* __restrict__ Bv,
    const int*   __restrict__ idx,
    bf16_t* __restrict__ Whi, bf16_t* __restrict__ Wlo)
{
    const int i = blockIdx.x * blockDim.x + threadIdx.x;  // 0..1023 (coalesced)
    const int o = blockIdx.y;                             // 0..3071
    const int b = blockIdx.z;                             // 0..7
    const int p = idx[b];

    float w = weight[(size_t)o * DIM + i];
    if (o < DIM) {
        const float* A = Aq + ((size_t)p * DIM + i) * RANK;
        const float* B = Bq + (size_t)p * RANK * DIM + o;   // broadcast across block
        float acc = 0.f;
#pragma unroll
        for (int r = 0; r < RANK; ++r) acc += A[r] * B[(size_t)r * DIM];
        w += acc;   // SCALE == 1.0
    } else if (o >= 2 * DIM) {
        const int j = o - 2 * DIM;
        const float* A = Av + ((size_t)p * DIM + i) * RANK;
        const float* B = Bv + (size_t)p * RANK * DIM + j;
        float acc = 0.f;
#pragma unroll
        for (int r = 0; r < RANK; ++r) acc += A[r] * B[(size_t)r * DIM];
        w += acc;
    }
    const bf16_t hi = (bf16_t)w;
    const bf16_t lo = (bf16_t)(w - (float)hi);
    const size_t base = ((size_t)b * ODIM + o) * DIM + i;
    Whi[base] = hi;
    Wlo[base] = lo;
}

// ---------------------------------------------------------------------------
// Kernel 2: batched GEMM  out[b] = x[b] @ W_eff[b]^T + bias
// bf16 hi/lo 3-product emulation on v_wmma_f32_16x16x32_bf16, f32 accumulate.
// Block = 128(M) x 128(O) tile, 256 threads = 8 waves (2 M-waves x 4 O-waves),
// wave sub-tile = 64x32 = 4x2 WMMA accumulators.
// Double-buffered LDS; W tiles staged via GLOBAL_LOAD_ASYNC_TO_LDS_B128
// (ASYNCcnt), x tiles VALU-converted f32->bf16 hi/lo.  One barrier per K-chunk.
// ---------------------------------------------------------------------------
__global__ void __launch_bounds__(256) gemm_qkv_kernel(
    const float*  __restrict__ x,
    const bf16_t* __restrict__ Whi,
    const bf16_t* __restrict__ Wlo,
    const float*  __restrict__ bias,
    float*        __restrict__ out)
{
    __shared__ __align__(16) bf16_t sX[2][2][TM * LDST];  // [buf][hi/lo]
    __shared__ __align__(16) bf16_t sW[2][2][TO * LDST];  // [buf][hi/lo]

    const int tid = threadIdx.x;
    const int o0  = blockIdx.x * TO;
    const int m0  = blockIdx.y * TM;
    const int b   = blockIdx.z;

    const float*  xb   = x   + (size_t)b * NROW * DIM;
    const bf16_t* whib = Whi + (size_t)b * ODIM * DIM;
    const bf16_t* wlob = Wlo + (size_t)b * ODIM * DIM;
    const uint64_t whib_u = (uint64_t)(uintptr_t)whib;   // block-uniform -> SGPR pair
    const uint64_t wlob_u = (uint64_t)(uintptr_t)wlob;

    const int wid   = tid >> 5;
    const int lane  = tid & 31;
    const int waveM = wid >> 2;                 // 0..1  -> 64 rows each
    const int waveO = wid & 3;                  // 0..3  -> 32 cols each
    const int lrow  = lane & 15;                // row/col within 16x16 tile
    const int khalf = (lane < 16) ? 0 : 8;      // ISA 16-bit A/B lane striping

    v8f acc[4][2];
#pragma unroll
    for (int t = 0; t < 4; ++t)
#pragma unroll
        for (int u = 0; u < 2; ++u)
#pragma unroll
            for (int r = 0; r < 8; ++r) acc[t][u][r] = 0.0f;

    // ---- stage x tile into buf: f32 -> bf16 hi/lo, 2 threads per row ----
    auto stage_x = [&](int buf, int kc) {
        const int row  = tid >> 1;
        const int half = (tid & 1) * 16;
        const float4* src = (const float4*)(xb + (size_t)(m0 + row) * DIM + kc + half);
        float f[16];
#pragma unroll
        for (int q = 0; q < 4; ++q) {
            float4 v = src[q];
            f[4*q+0] = v.x; f[4*q+1] = v.y; f[4*q+2] = v.z; f[4*q+3] = v.w;
        }
        v8bf h0, h1, l0, l1;
#pragma unroll
        for (int q = 0; q < 8; ++q) {
            bf16_t ha = (bf16_t)f[q];
            bf16_t hb = (bf16_t)f[8 + q];
            h0[q] = ha;  l0[q] = (bf16_t)(f[q]     - (float)ha);
            h1[q] = hb;  l1[q] = (bf16_t)(f[8 + q] - (float)hb);
        }
        bf16_t* dh = &sX[buf][0][row * LDST + half];
        bf16_t* dl = &sX[buf][1][row * LDST + half];
        *(v8bf*)dh = h0;  *(v8bf*)(dh + 8) = h1;
        *(v8bf*)dl = l0;  *(v8bf*)(dl + 8) = l1;
    };

    // ---- stage W tiles into buf via async DMA: thread t -> row t/2, 32B seg ----
    auto issue_w = [&](int buf, int kc) {
        const int row      = tid >> 1;
        const uint32_t seg = (uint32_t)(tid & 1) * 32;             // byte segment in 64B row
        const uint32_t gof = ((uint32_t)(o0 + row) * DIM + (uint32_t)kc) * 2 + seg;
        const uint32_t lhi = lds_off(&sW[buf][0][row * LDST]) + seg;
        const uint32_t llo = lds_off(&sW[buf][1][row * LDST]) + seg;
        async_ld_b128(lhi,      whib_u, gof);
        async_ld_b128(lhi + 16, whib_u, gof + 16);
        async_ld_b128(llo,      wlob_u, gof);
        async_ld_b128(llo + 16, wlob_u, gof + 16);
    };

    // ---- fragment loads + 3-product split-precision WMMA on buf ----
    auto compute = [&](int buf) {
        v16bf ahi[4], alo[4], bhi[2], blo[2];
#pragma unroll
        for (int t = 0; t < 4; ++t) {
            const int row = waveM * 64 + t * 16 + lrow;
            const bf16_t* ph = &sX[buf][0][row * LDST + khalf];
            const bf16_t* pl = &sX[buf][1][row * LDST + khalf];
            v8bf h0 = *(const v8bf*)ph, h1 = *(const v8bf*)(ph + 16);
            v8bf l0 = *(const v8bf*)pl, l1 = *(const v8bf*)(pl + 16);
#pragma unroll
            for (int q = 0; q < 8; ++q) {
                ahi[t][q] = h0[q]; ahi[t][8 + q] = h1[q];
                alo[t][q] = l0[q]; alo[t][8 + q] = l1[q];
            }
        }
#pragma unroll
        for (int u = 0; u < 2; ++u) {
            const int orow = waveO * 32 + u * 16 + lrow;   // W row o == B column
            const bf16_t* ph = &sW[buf][0][orow * LDST + khalf];
            const bf16_t* pl = &sW[buf][1][orow * LDST + khalf];
            v8bf h0 = *(const v8bf*)ph, h1 = *(const v8bf*)(ph + 16);
            v8bf l0 = *(const v8bf*)pl, l1 = *(const v8bf*)(pl + 16);
#pragma unroll
            for (int q = 0; q < 8; ++q) {
                bhi[u][q] = h0[q]; bhi[u][8 + q] = h1[q];
                blo[u][q] = l0[q]; blo[u][8 + q] = l1[q];
            }
        }
#pragma unroll
        for (int t = 0; t < 4; ++t)
#pragma unroll
            for (int u = 0; u < 2; ++u) {
                acc[t][u] = __builtin_amdgcn_wmma_f32_16x16x32_bf16(
                    false, ahi[t], false, bhi[u], (short)0, acc[t][u], false, false);
                acc[t][u] = __builtin_amdgcn_wmma_f32_16x16x32_bf16(
                    false, ahi[t], false, blo[u], (short)0, acc[t][u], false, false);
                acc[t][u] = __builtin_amdgcn_wmma_f32_16x16x32_bf16(
                    false, alo[t], false, bhi[u], (short)0, acc[t][u], false, false);
            }
    };

    // ---- pipelined main loop: one barrier per K-chunk ----
    stage_x(0, 0);
    issue_w(0, 0);
    for (int it = 0; it < NIT; ++it) {
        const int cur = it & 1;
        wait_async0();        // own wave's W DMA for `cur` complete (LDS written)
        __syncthreads();      // all waves' x stores + W DMA for `cur` complete;
                              // also gates WAR on buffer `cur^1` rewritten below
        if (it + 1 < NIT) {
            stage_x(cur ^ 1, (it + 1) * TK);
            issue_w(cur ^ 1, (it + 1) * TK);
        }
        compute(cur);
    }

    // ---- epilogue: C layout VGPR r -> M = r (+8 for lanes 16..31), N = lane%16 ----
#pragma unroll
    for (int u = 0; u < 2; ++u) {
        const int o  = o0 + waveO * 32 + u * 16 + lrow;
        const float bv = bias[o];
#pragma unroll
        for (int t = 0; t < 4; ++t) {
            const int mbase = m0 + waveM * 64 + t * 16 + ((lane < 16) ? 0 : 8);
#pragma unroll
            for (int r = 0; r < 8; ++r)
                out[((size_t)b * NROW + mbase + r) * ODIM + o] = acc[t][u][r] + bv;
        }
    }
}

// ---------------------------------------------------------------------------
extern "C" void kernel_launch(void* const* d_in, const int* in_sizes, int n_in,
                              void* d_out, int out_size, void* d_ws, size_t ws_size,
                              hipStream_t stream)
{
    (void)in_sizes; (void)n_in; (void)out_size; (void)ws_size;
    const float* x      = (const float*)d_in[0];
    const float* weight = (const float*)d_in[1];
    const float* bias   = (const float*)d_in[2];
    const float* Aq     = (const float*)d_in[3];
    const float* Bq     = (const float*)d_in[4];
    const float* Av     = (const float*)d_in[5];
    const float* Bv     = (const float*)d_in[6];
    const int*   idx    = (const int*)d_in[7];
    // d_in[8] = frozen_mask: only affects gradients (stop_gradient); forward-invariant.
    float* out = (float*)d_out;

    bf16_t* Whi = (bf16_t*)d_ws;                       // 8*3072*1024*2 B = 50.3 MB
    bf16_t* Wlo = Whi + (size_t)BATCH * ODIM * DIM;    // +50.3 MB

    dim3 g1(DIM / 256, ODIM, BATCH);
    merge_w_kernel<<<g1, 256, 0, stream>>>(weight, Aq, Bq, Av, Bv, idx, Whi, Wlo);

    dim3 g2(ODIM / TO, NROW / TM, BATCH);
    gemm_qkv_kernel<<<g2, 256, 0, stream>>>(x, Whi, Wlo, bias, out);
}
